// NTXentLoss_39986145526198
// MI455X (gfx1250) — matrix operand order repaired and checked by reading, not compiled, not run on hardware
//
#include <hip/hip_runtime.h>

// ---------------------------------------------------------------------------
// NT-Xent loss, fused for MI455X (gfx1250, wave32, WMMA bf16 16x16x32).
//   B=4096, D=256, M=2B=8192, T=0.5 (invT = 2.0)
// loss = mean_i [ log( S_i - exp(d_ii/T) ) - p_i/T ]
//   S_i  = sum_j exp(sim_ij/T)   (all j, diagonal removed via d_ii)
//   p_i  = dot(zn_i, zn_{(i+4096)%8192})
// ---------------------------------------------------------------------------

typedef __attribute__((ext_vector_type(16))) __bf16        v16bf;
typedef __attribute__((ext_vector_type(8)))  float         v8f;
typedef __attribute__((ext_vector_type(4)))  unsigned int  u32x4;

union ABFrag { u32x4 q[2]; v16bf v; };

#define MROWS 8192
#define DK    256
#define LDSP  264   // 256 halfs + 8 pad -> 528B row stride, kills bank conflicts

// exp(x/T) = exp2(x * invT * log2(e)); invT = 2.0
#define EXP2_SCALE 2.8853900817779268f

// ---------------- Kernel 1: normalize rows of concat(zjs, zis) -> bf16 ------
__global__ void ntxent_normalize(const float* __restrict__ zis,
                                 const float* __restrict__ zjs,
                                 unsigned short* __restrict__ zn) {
  const int row = blockIdx.x;   // 0..8191 ; rows 0..4095 = zjs, rest = zis
  const int t   = threadIdx.x;  // 0..255 (one element each)
  const float* src = (row < 4096) ? (zjs + (size_t)row * DK)
                                  : (zis + (size_t)(row - 4096) * DK);
  float v = src[t];
  __shared__ float red[256];
  red[t] = v * v;
  __syncthreads();
  for (int s = 128; s > 0; s >>= 1) {
    if (t < s) red[t] += red[t + s];
    __syncthreads();
  }
  float norm = fmaxf(sqrtf(red[0]), 1e-8f);
  float z = v / norm;
  // f32 -> bf16 round-to-nearest-even
  unsigned int u = __float_as_uint(z);
  u += 0x7FFFu + ((u >> 16) & 1u);
  zn[(size_t)row * DK + t] = (unsigned short)(u >> 16);
}

// ---------------- Kernel 2: fused Gram + exp-sum (the WMMA kernel) ----------
// Grid: 512 blocks = 64 row-blocks (128 rows each) x 8 column chunks (1024 cols).
// Block: 256 threads = 8 waves; wave w owns rows [rowblock*128 + w*16, +16).
__global__ __launch_bounds__(256) void ntxent_simexp(
    const unsigned short* __restrict__ zn,
    float* __restrict__ partial /* [8192][8] */) {
  __shared__ unsigned short ldsB[2][16][LDSP];   // 16.5 KB double-buffered B panel

  const int tid      = threadIdx.x;
  const int lane     = tid & 31;
  const int wave     = tid >> 5;          // 0..7
  const int rowblock = blockIdx.x & 63;   // 0..63
  const int colchunk = blockIdx.x >> 6;   // 0..7
  const int rowbase  = rowblock * 128 + wave * 16;
  const int m  = lane & 15;               // row-in-tile (A) / col-in-tile (B)
  const int hi = lane >> 4;               // K-half selector

  // A panel: 16 rows x 256 K, resident in registers (8 frags x 8 VGPRs).
  // WMMA 16-bit A layout: lane m   -> halfs {K0..K0+7, K0+16..K0+23}
  //                       lane m+16-> halfs {K0+8..K0+15, K0+24..K0+31}
  ABFrag A[8];
  {
    const u32x4* rowp = (const u32x4*)(zn + (size_t)(rowbase + m) * DK);
    #pragma unroll
    for (int ks = 0; ks < 8; ++ks) {
      A[ks].q[0] = rowp[4 * ks + hi];
      A[ks].q[1] = rowp[4 * ks + 2 + hi];
    }
  }

  float sacc[8];
  #pragma unroll
  for (int r = 0; r < 8; ++r) sacc[r] = 0.0f;

  const int c0base = colchunk * 1024;

  // Staging geometry: 256 threads move one 16-col x 256-K bf16 panel (8 KB).
  const int sn  = tid >> 4;   // 0..15 column within tile
  const int sch = tid & 15;   // 0..15 chunk of 16 halfs (32 B)

  // Prologue: stage tile 0 into buffer 0.
  {
    const u32x4* src = (const u32x4*)(zn + (size_t)(c0base + sn) * DK + sch * 16);
    u32x4 a = src[0];
    u32x4 b = src[1];
    u32x4* dst = (u32x4*)&ldsB[0][sn][sch * 16];
    dst[0] = a;
    dst[1] = b;
  }
  __syncthreads();

  for (int ct = 0; ct < 64; ++ct) {
    const int buf = ct & 1;
    // Branchless tail clamp: ct+1 = 64 -> restage tile 63 into the dead buffer.
    const int nt = (ct + 1) - ((ct + 1) >> 6);

    // 1) Preload ALL B fragments for this tile; the asm fence forces all 16
    //    ds_load_b128 to complete (one dscnt wait) before any WMMA issues.
    //    WMMA 16-bit B layout: lane n needs z[col n][k0..k0+7, k0+16..k0+23],
    //                          lane n+16 needs z[col n][k0+8..k0+15, k0+24..k0+31].
    ABFrag Bf[8];
    #pragma unroll
    for (int ks = 0; ks < 8; ++ks) {
      const u32x4* bp = (const u32x4*)&ldsB[buf][m][ks * 32];
      Bf[ks].q[0] = bp[hi];
      Bf[ks].q[1] = bp[2 + hi];
    }
    #pragma unroll
    for (int ks = 0; ks < 8; ++ks)
      asm volatile("" : "+v"(Bf[ks].q[0]), "+v"(Bf[ks].q[1]));

    // 2) Issue staging global loads now; latency hides under the WMMA chain.
    const u32x4* ssrc =
        (const u32x4*)(zn + (size_t)(c0base + nt * 16 + sn) * DK + sch * 16);
    u32x4 sa = ssrc[0];
    u32x4 sb = ssrc[1];

    // 3) 16x16 similarity tile, K=256 in 8 WMMA steps; two accumulator
    //    chains (even/odd K) to break the serial C dependency.
    v8f C0 = {};
    v8f C1 = {};
    #pragma unroll
    for (int ks = 0; ks < 8; ks += 2) {
      C0 = __builtin_amdgcn_wmma_f32_16x16x32_bf16(
               false, A[ks].v,     false, Bf[ks].v,     (short)0, C0, false, false);
      C1 = __builtin_amdgcn_wmma_f32_16x16x32_bf16(
               false, A[ks + 1].v, false, Bf[ks + 1].v, (short)0, C1, false, false);
    }

    // 4) Staging stores after the WMMAs: the pre-WMMA dscnt wait covers only
    //    the B loads; the end-of-loop barrier covers these stores.
    {
      u32x4* dst = (u32x4*)&ldsB[buf ^ 1][sn][sch * 16];
      dst[0] = sa;
      dst[1] = sb;
    }

    // 5) exp accumulate. C layout: VGPR r, lanes 0-15 -> M=r,
    //    lanes 16-31 -> M=8+r, N=lane%16. exp(sim/T) <= e^2: no max needed.
    #pragma unroll
    for (int r = 0; r < 8; ++r)
      sacc[r] += __builtin_amdgcn_exp2f((C0[r] + C1[r]) * EXP2_SCALE);

    __syncthreads();
  }

  // Sum over the 16 lanes that share a row half.
  #pragma unroll
  for (int r = 0; r < 8; ++r) {
    float s = sacc[r];
    s += __shfl_xor(s, 1);
    s += __shfl_xor(s, 2);
    s += __shfl_xor(s, 4);
    s += __shfl_xor(s, 8);
    sacc[r] = s;
  }
  if (lane == 0 || lane == 16) {
    const int mbase = rowbase + hi * 8;
    #pragma unroll
    for (int r = 0; r < 8; ++r)
      partial[(size_t)(mbase + r) * 8 + colchunk] = sacc[r];
  }
}

// ---------------- Kernel 3: per-row pos/diag dot + loss ---------------------
__global__ void ntxent_rowloss(const unsigned short* __restrict__ zn,
                               const float* __restrict__ partial,
                               float* __restrict__ loss) {
  const int lane = threadIdx.x & 31;
  const int wave = threadIdx.x >> 5;
  const int row  = blockIdx.x * 8 + wave;
  const int prow = (row + 4096) & (MROWS - 1);
  const unsigned short* a = zn + (size_t)row  * DK;
  const unsigned short* b = zn + (size_t)prow * DK;
  float dp = 0.f, dd = 0.f;
  #pragma unroll
  for (int e = 0; e < 8; ++e) {
    const int idx = lane * 8 + e;
    const float av = __uint_as_float(((unsigned int)a[idx]) << 16);
    const float bv = __uint_as_float(((unsigned int)b[idx]) << 16);
    dp += av * bv;
    dd += av * av;
  }
  #pragma unroll
  for (int off = 1; off < 32; off <<= 1) {
    dp += __shfl_xor(dp, off);
    dd += __shfl_xor(dd, off);
  }
  if (lane == 0) {
    float S = 0.f;
    #pragma unroll
    for (int c = 0; c < 8; ++c) S += partial[(size_t)row * 8 + c];
    loss[row] = logf(S - __builtin_amdgcn_exp2f(dd * EXP2_SCALE)) - dp * 2.0f;
  }
}

// ---------------- Kernel 4: mean over 8192 rows -----------------------------
__global__ void ntxent_reduce(const float* __restrict__ loss,
                              float* __restrict__ out) {
  __shared__ float red[256];
  const int t = threadIdx.x;
  float s = 0.f;
  for (int i = t; i < MROWS; i += 256) s += loss[i];
  red[t] = s;
  __syncthreads();
  for (int k = 128; k > 0; k >>= 1) {
    if (t < k) red[t] += red[t + k];
    __syncthreads();
  }
  if (t == 0) out[0] = red[0] * (1.0f / (float)MROWS);
}

// ---------------------------------------------------------------------------
extern "C" void kernel_launch(void* const* d_in, const int* in_sizes, int n_in,
                              void* d_out, int out_size, void* d_ws, size_t ws_size,
                              hipStream_t stream) {
  const float* zis = (const float*)d_in[0];
  const float* zjs = (const float*)d_in[1];

  unsigned short* zn = (unsigned short*)d_ws;                       // 4 MB bf16 [8192][256]
  float* partial = (float*)((char*)d_ws + (size_t)MROWS * DK * 2);  // 256 KB [8192][8]
  float* loss    = partial + (size_t)MROWS * 8;                     // 32 KB  [8192]
  float* out     = (float*)d_out;

  ntxent_normalize<<<MROWS, 256, 0, stream>>>(zis, zjs, zn);
  ntxent_simexp  <<<512,   256, 0, stream>>>(zn, partial);
  ntxent_rowloss <<<1024,  256, 0, stream>>>(zn, partial, loss);
  ntxent_reduce  <<<1,     256, 0, stream>>>(loss, out);
}
